// RoIWeightedSumLayer_33827162423849
// MI455X (gfx1250) — matrix operand order repaired and checked by reading, not compiled, not run on hardware
//
#include <hip/hip_runtime.h>
#include <hip/hip_bf16.h>
#include <cstdint>

// ---------------------------------------------------------------------------
// RoI softmax-weighted pooling as a WMMA GEMM on gfx1250 (MI455X).
//   out[r,c] = sum_p w[r,p] * input[bid[r],c,p],  w = box-masked softmax(score)
// w[r,p] = exp(s[p]) * scale[r],  scale[r] = exp(-max_r)/denom_r, so the hot
// loop is A = expmap*scale (f16), B = input (f16), C += A*B via
// v_wmma_f32_16x16x32_f16 (f32 accumulate). All fragment loads are
// unconditional 128-bit loads (masking is branchless cndmask on A only);
// out-of-window pixels are provably A==0, so B may load don't-care data as
// long as addresses stay in-bounds (guaranteed by 32-aligned chunk start +
// linear-index clamp to HW-4). Per-tile union boxes are precomputed so the
// GEMM kernel's control flow is pure SALU. input (82 MB) fits in 192 MB L2.
// ---------------------------------------------------------------------------

#define SPS     0.25f
#define N_IMG   2
#define C_CH    256
#define H_DIM   200
#define W_DIM   200
#define R_ROIS  512
#define HW      (H_DIM * W_DIM)
#define Z_SPLIT 4
#define N_TILES (R_ROIS / 16)

typedef __attribute__((ext_vector_type(16))) _Float16 v16h;
typedef __attribute__((ext_vector_type(8)))  float    v8f;

__device__ __forceinline__ float4 ld4_clamped(const float* __restrict__ plane, int idx) {
  idx = min(idx, HW - 4);                       // only clamps where A==0
  return *reinterpret_cast<const float4*>(plane + idx);
}

// ---- Stage 0: exp(score) map ----------------------------------------------
__global__ void expmap_kernel(const float* __restrict__ score,
                              float* __restrict__ expmap, int n) {
  int i = blockIdx.x * blockDim.x + threadIdx.x;
  if (i < n) expmap[i] = expf(score[i]);
}

__global__ void zero_kernel(float* __restrict__ p, int n) {
  int i = blockIdx.x * blockDim.x + threadIdx.x;
  if (i < n) p[i] = 0.0f;
}

// ---- Stage 1: per-RoI box + softmax stats (one 256-thread block per RoI) --
__global__ __launch_bounds__(256)
void roi_stats_kernel(const float* __restrict__ rois,
                      const float* __restrict__ score,
                      int* __restrict__ boxOut,     // [R*4] x1,y1,x2,y2
                      int* __restrict__ bidOut,     // [R]
                      float* __restrict__ scaleOut) // [R] exp(-m)/denom (0 if empty)
{
  __shared__ float red[256];
  const int r = blockIdx.x;
  const int tid = threadIdx.x;
  const float* roi = rois + r * 5;
  int bid = (int)roi[0];
  int x1 = (int)roundf(roi[1] * SPS);
  int y1 = (int)roundf(roi[2] * SPS);
  int x2 = (int)roundf(roi[3] * SPS);
  int y2 = (int)roundf(roi[4] * SPS);
  x1 = max(0, min(x1, W_DIM)); x2 = max(0, min(x2, W_DIM));
  y1 = max(0, min(y1, H_DIM)); y2 = max(0, min(y2, H_DIM));
  const int uw = x2 - x1, uh = y2 - y1;
  const int area = (uw > 0 && uh > 0) ? uw * uh : 0;
  const float* s = score + bid * HW;

  // pass 1: max over box
  float lm = -3.0e38f;
  for (int i = tid; i < area; i += 256) {
    int py = y1 + i / uw, px = x1 + i % uw;
    lm = fmaxf(lm, s[py * W_DIM + px]);
  }
  red[tid] = lm; __syncthreads();
  for (int st = 128; st > 0; st >>= 1) {
    if (tid < st) red[tid] = fmaxf(red[tid], red[tid + st]);
    __syncthreads();
  }
  const float m = red[0]; __syncthreads();

  // pass 2: denom = sum exp(s - m)
  float ls = 0.0f;
  for (int i = tid; i < area; i += 256) {
    int py = y1 + i / uw, px = x1 + i % uw;
    ls += expf(s[py * W_DIM + px] - m);
  }
  red[tid] = ls; __syncthreads();
  for (int st = 128; st > 0; st >>= 1) {
    if (tid < st) red[tid] += red[tid + st];
    __syncthreads();
  }
  const float denom = red[0];

  if (tid == 0) {
    boxOut[r * 4 + 0] = x1; boxOut[r * 4 + 1] = y1;
    boxOut[r * 4 + 2] = x2; boxOut[r * 4 + 3] = y2;
    bidOut[r] = bid;
    float sc = 0.0f;
    if (area > 0 && denom > 0.0f) sc = expf(-m) / fmaxf(denom, 1e-30f);
    scaleOut[r] = sc;
  }
}

// ---- Stage 1b: union box per (16-RoI tile, image) -------------------------
// uni[(tile*N_IMG + n)*4 + {0..3}] = {ux1,uy1,ux2,uy2} (ux2<=ux1 -> empty)
__global__ __launch_bounds__(64)
void tile_union_kernel(const int* __restrict__ box, const int* __restrict__ bid,
                       int* __restrict__ uni) {
  int t = blockIdx.x * blockDim.x + threadIdx.x;
  if (t >= N_TILES * N_IMG) return;
  const int tile = t / N_IMG, n = t % N_IMG;
  int ux1 = W_DIM, uy1 = H_DIM, ux2 = 0, uy2 = 0;
  for (int q = 0; q < 16; ++q) {
    int r = tile * 16 + q;
    int x1 = box[r * 4 + 0], y1 = box[r * 4 + 1];
    int x2 = box[r * 4 + 2], y2 = box[r * 4 + 3];
    if (bid[r] == n && x2 > x1 && y2 > y1) {
      ux1 = min(ux1, x1); uy1 = min(uy1, y1);
      ux2 = max(ux2, x2); uy2 = max(uy2, y2);
    }
  }
  uni[t * 4 + 0] = ux1; uni[t * 4 + 1] = uy1;
  uni[t * 4 + 2] = ux2; uni[t * 4 + 3] = uy2;
}

// ---- Stage 2: WMMA weighted-sum GEMM --------------------------------------
// Tile: 16 RoIs (block.x) x 64 channels (block.y, 16 per wave), K split over
// union-box rows by block.z. Each wave owns one 16x16 f32 accumulator.
__global__ __launch_bounds__(128)
void roi_wsum_wmma(const float* __restrict__ input,   // [N,C,H,W] f32
                   const float* __restrict__ expmap,  // [N,H,W]   f32
                   const int* __restrict__ box,
                   const int* __restrict__ bid,
                   const float* __restrict__ scale,
                   const int* __restrict__ uni,       // [N_TILES*N_IMG*4]
                   float* __restrict__ out)           // [R,C] f32 (pre-zeroed)
{
  const int tid  = threadIdx.x;
  const int lane = tid & 31;
  const int wave = tid >> 5;
  const int r0 = blockIdx.x * 16;
  const int c0 = blockIdx.y * 64 + wave * 16;
  const int zz = blockIdx.z;

  const int mRow = lane & 15;   // A row / output row (within tile)
  const int hi   = lane >> 4;   // lane half selector
  const int nCol = lane & 15;   // B col / output col (within tile)

  // per-lane RoI metadata (divergent loads, done once)
  const int bx1 = box[(r0 + mRow) * 4 + 0], by1 = box[(r0 + mRow) * 4 + 1];
  const int bx2 = box[(r0 + mRow) * 4 + 2], by2 = box[(r0 + mRow) * 4 + 3];
  const int   myBid = bid[r0 + mRow];
  const float mySc  = scale[r0 + mRow];

  v8f acc = {};

  for (int n = 0; n < N_IMG; ++n) {
    // precomputed wave-uniform union box -> uniform scalar loads + SALU test
    const int ub = (blockIdx.x * N_IMG + n) * 4;
    const int ux1 = uni[ub + 0], uy1 = uni[ub + 1];
    const int ux2 = uni[ub + 2], uy2 = uni[ub + 3];
    if (ux2 <= ux1 || uy2 <= uy1) continue;       // uniform: EXEC stays full
    const int uh = uy2 - uy1;
    const int ys = uy1 + (uh * zz) / Z_SPLIT;     // this block's K slice (rows)
    const int ye = uy1 + (uh * (zz + 1)) / Z_SPLIT;
    const int pxStart = ux1 & ~31;                // 32-aligned -> 16B-aligned float4s

    const float* emap  = expmap + n * HW;                          // image plane
    const float* chanB = input + ((n * C_CH) + c0 + nCol) * HW;    // lane's channel plane
    const bool  rowOk = (myBid == n) && (bx2 > bx1) && (by2 > by1);

    for (int py = ys; py < ye; ++py) {
      const float rowSc = (rowOk && py >= by1 && py < by2) ? mySc : 0.0f;
      // prefetch next row's data (global_prefetch_b8)
      if (py + 1 < ye) {
        __builtin_prefetch(chanB + (py + 1) * W_DIM + pxStart, 0, 3);
        __builtin_prefetch(emap  + (py + 1) * W_DIM + pxStart, 0, 3);
      }
      for (int pxB = pxStart; pxB < ux2; pxB += 32) {
        // ---- B fragment: 16 contiguous pixels per lane, 4x b128 loads -----
        const int pxBB = pxB + hi * 16;
        float4 b0 = ld4_clamped(chanB, py * W_DIM + pxBB + 0);
        float4 b1 = ld4_clamped(chanB, py * W_DIM + pxBB + 4);
        float4 b2 = ld4_clamped(chanB, py * W_DIM + pxBB + 8);
        float4 b3 = ld4_clamped(chanB, py * W_DIM + pxBB + 12);
        // ---- A fragment: two 8-pixel runs per lane, 4x b128 loads ---------
        const int pxA0 = pxB + hi * 16;        // run0: K = hi*16 + 0..7
        const int pxA1 = pxB + hi * 16 + 8;    // run1: K = hi*16 + 8..15
        float4 a0 = ld4_clamped(emap, py * W_DIM + pxA0 + 0);
        float4 a1 = ld4_clamped(emap, py * W_DIM + pxA0 + 4);
        float4 a2 = ld4_clamped(emap, py * W_DIM + pxA1 + 0);
        float4 a3 = ld4_clamped(emap, py * W_DIM + pxA1 + 4);

        // branchless mask: g = rowSc inside [bx1,bx2), else 0
        float aw[16] = { a0.x, a0.y, a0.z, a0.w, a1.x, a1.y, a1.z, a1.w,
                         a2.x, a2.y, a2.z, a2.w, a3.x, a3.y, a3.z, a3.w };
        v16h a, b;
        #pragma unroll
        for (int j = 0; j < 16; ++j) {
          int px = (j < 8) ? (pxA0 + j) : (pxA1 + (j - 8));
          float g = (px >= bx1 && px < bx2) ? rowSc : 0.0f;
          a[j] = (_Float16)(aw[j] * g);
        }
        float bw[16] = { b0.x, b0.y, b0.z, b0.w, b1.x, b1.y, b1.z, b1.w,
                         b2.x, b2.y, b2.z, b2.w, b3.x, b3.y, b3.z, b3.w };
        #pragma unroll
        for (int j = 0; j < 16; ++j) b[j] = (_Float16)bw[j];

        acc = __builtin_amdgcn_wmma_f32_16x16x32_f16(
            /*neg_a=*/false, a, /*neg_b=*/false, b,
            /*c_mod=*/(short)0, acc, /*reuse_a=*/false, /*reuse_b=*/false);
      }
    }
  }

  // D layout (32-bit 16x16): VGPR v, lanes0-15 -> M=v, lanes16-31 -> M=v+8; N=lane&15
  #pragma unroll
  for (int v = 0; v < 8; ++v) {
    int row = r0 + v + hi * 8;
    int col = c0 + nCol;
    atomicAdd(&out[row * C_CH + col], acc[v]);
  }
}

// ---------------------------------------------------------------------------
extern "C" void kernel_launch(void* const* d_in, const int* in_sizes, int n_in,
                              void* d_out, int out_size, void* d_ws, size_t ws_size,
                              hipStream_t stream) {
  const float* input = (const float*)d_in[0];   // [2,256,200,200]
  const float* rois  = (const float*)d_in[1];   // [512,5]
  const float* score = (const float*)d_in[2];   // [2,1,200,200]
  float* out = (float*)d_out;                   // [512,256]

  char*  ws     = (char*)d_ws;
  float* expmap = (float*)ws;                                      // N*HW floats
  int*   boxB   = (int*)(ws + (size_t)N_IMG * HW * sizeof(float)); // R*4 ints
  int*   bidB   = boxB + R_ROIS * 4;                               // R ints
  float* scaleB = (float*)(bidB + R_ROIS);                         // R floats
  int*   uniB   = (int*)(scaleB + R_ROIS);                         // N_TILES*N_IMG*4 ints

  expmap_kernel<<<(N_IMG * HW + 255) / 256, 256, 0, stream>>>(score, expmap, N_IMG * HW);
  roi_stats_kernel<<<R_ROIS, 256, 0, stream>>>(rois, score, boxB, bidB, scaleB);
  tile_union_kernel<<<1, 64, 0, stream>>>(boxB, bidB, uniB);
  zero_kernel<<<(R_ROIS * C_CH + 255) / 256, 256, 0, stream>>>(out, R_ROIS * C_CH);
  roi_wsum_wmma<<<dim3(N_TILES, C_CH / 64, Z_SPLIT), 128, 0, stream>>>(
      input, expmap, boxB, bidB, scaleB, uniB, out);
}
